// RelativeMultiHeadAttention_27393301413964
// MI455X (gfx1250) — matrix operand order repaired and checked by reading, not compiled, not run on hardware
//
#include <hip/hip_runtime.h>
#include <hip/hip_bf16.h>
#include <hip/hip_fp16.h>

#define NHEADS 16
#define HDIM   64
#define SEQ    1024
#define NB     4
#define DMODEL 1024
#define MAXREL 64
#define WINDOW 256
#define RELPAD 144   // 129 rel rows padded to 9*16
#define KWIN   352   // max staged key-window per 64-row query block (32-aligned)

typedef __attribute__((ext_vector_type(16))) _Float16 v16h;
typedef __attribute__((ext_vector_type(8)))  _Float16 v8h;
typedef __attribute__((ext_vector_type(8)))  float    v8f;

// ---------------- WMMA fragment helpers (wave32, 16x16x32 f16) ----------------
// A (16x32): lane row = lane&15; lanes<16 hold K{0..7,16..23}, lanes>=16 K{8..15,24..31}
static __device__ __forceinline__ v16h load_a_frag(const _Float16* A, int lda) {
    const int lane = threadIdx.x & 31;
    const int r    = lane & 15;
    const int kb   = (lane >> 4) << 3;      // 0 or 8
    const _Float16* p = A + r * lda + kb;
    v8h lo = *(const v8h*)(p);
    v8h hi = *(const v8h*)(p + 16);
    v16h out;
#pragma unroll
    for (int i = 0; i < 8; ++i) { out[i] = lo[i]; out[i + 8] = hi[i]; }
    return out;
}

// B (32x16): lane col = lane&15; lanes<16 hold K{0..7,16..23}, lanes>=16 K{8..15,24..31}.
// Source must be stored [N][K] row-major (i.e. the transpose of the math B).
static __device__ __forceinline__ v16h load_b_frag(const _Float16* Bt, int ldb) {
    const int lane = threadIdx.x & 31;
    const int n    = lane & 15;
    const int kb   = (lane >> 4) << 3;      // 0 or 8
    const _Float16* p = Bt + n * ldb + kb;
    v8h lo = *(const v8h*)(p);
    v8h hi = *(const v8h*)(p + 16);
    v16h out;
#pragma unroll
    for (int i = 0; i < 8; ++i) { out[i] = lo[i]; out[i + 8] = hi[i]; }
    return out;
}

static __device__ __forceinline__ v8f wmma_f16(v16h a, v16h b, v8f c) {
    return __builtin_amdgcn_wmma_f32_16x16x32_f16(false, a, false, b, (short)0, c,
                                                  false, false);
}

// ---------------- conversion kernels ----------------
__global__ void k_cvt_f16(const float* __restrict__ s, _Float16* __restrict__ d, int n) {
    int i = blockIdx.x * blockDim.x + threadIdx.x;
    if (i < n) d[i] = (_Float16)s[i];
}

__global__ void k_rel_pad(const float* __restrict__ s, _Float16* __restrict__ d) {
    int i = blockIdx.x * blockDim.x + threadIdx.x;      // RELPAD*HDIM
    if (i < RELPAD * HDIM) {
        int r = i >> 6;
        d[i] = (r < 2 * MAXREL + 1) ? (_Float16)s[i] : (_Float16)0.0f;
    }
}

// ---------------- QKV projection GEMM ----------------
// C[m][n] = sum_k Xh[m][k] * Wqkv[n][k];  M=4096, N=3072, K=1024
// scatter: n -> (which, head, d).  Q,K stored [bh][l][d]; V stored transposed [bh][d][l]
__global__ __launch_bounds__(256)
void k_qkv_gemm(const _Float16* __restrict__ Xh, const _Float16* __restrict__ Wh,
                _Float16* __restrict__ Qf, _Float16* __restrict__ Kf,
                _Float16* __restrict__ Vt) {
    const int lane = threadIdx.x & 31;
    const int wave = threadIdx.x >> 5;          // 0..7 -> 2(M) x 4(N)
    const int mw = wave & 1, nw = wave >> 1;
    const int m0 = blockIdx.y * 64 + mw * 32;
    const int n0 = blockIdx.x * 128 + nw * 32;
    const _Float16* Ab = Xh + m0 * DMODEL;
    const _Float16* Bb = Wh + n0 * DMODEL;

    v8f acc[2][2] = {};
    for (int k = 0; k < DMODEL; k += 32) {
        if (k + 32 < DMODEL) {
            __builtin_prefetch(Ab + k + 32, 0, 1);
            __builtin_prefetch(Bb + k + 32, 0, 1);
        }
        v16h a0 = load_a_frag(Ab + k, DMODEL);
        v16h a1 = load_a_frag(Ab + 16 * DMODEL + k, DMODEL);
        v16h b0 = load_b_frag(Bb + k, DMODEL);
        v16h b1 = load_b_frag(Bb + 16 * DMODEL + k, DMODEL);
        acc[0][0] = wmma_f16(a0, b0, acc[0][0]);
        acc[0][1] = wmma_f16(a0, b1, acc[0][1]);
        acc[1][0] = wmma_f16(a1, b0, acc[1][0]);
        acc[1][1] = wmma_f16(a1, b1, acc[1][1]);
    }

    const int half8 = (lane >> 4) << 3;
    const int nn = lane & 15;
#pragma unroll
    for (int mi = 0; mi < 2; ++mi)
#pragma unroll
        for (int ni = 0; ni < 2; ++ni)
#pragma unroll
            for (int v = 0; v < 8; ++v) {
                const int m = m0 + mi * 16 + half8 + v;
                const int n = n0 + ni * 16 + nn;
                const _Float16 val = (_Float16)acc[mi][ni][v];
                const int which = n >> 10;
                const int hh = (n & 1023) >> 6;
                const int dd = n & 63;
                const int bb = m >> 10;
                const int lr = m & 1023;
                const int bh = bb * NHEADS + hh;
                if (which == 0)      Qf[(bh * SEQ + lr) * HDIM + dd] = val;
                else if (which == 1) Kf[(bh * SEQ + lr) * HDIM + dd] = val;
                else                 Vt[(bh * HDIM + dd) * SEQ + lr] = val;
            }
}

// ---------------- windowed relative attention (flash-style) ----------------
// block = 4 waves, 64 query rows; the block's shared K/V window is staged into
// LDS with GLOBAL_LOAD_ASYNC_TO_LDS (ASYNCcnt), then each wave streams 32-key
// tiles out of LDS through WMMA with online softmax.
__global__ __launch_bounds__(128)
void k_attn(const _Float16* __restrict__ Qf, const _Float16* __restrict__ Kf,
            const _Float16* __restrict__ Vt, const _Float16* __restrict__ Relh,
            _Float16* __restrict__ Oh) {
    __shared__ float    qE[64 * RELPAD];        // 36.9 KB
    __shared__ _Float16 Ks[KWIN * HDIM];        // 44 KB   [j][d]
    __shared__ _Float16 Vs[HDIM * KWIN];        // 44 KB   [d][j]
    __shared__ _Float16 pbuf[4 * 16 * 32];      // 4 KB    per-wave P staging

    const int lane = threadIdx.x & 31;
    const int wave = threadIdx.x >> 5;          // 0..3
    const int bh = blockIdx.y;
    const int b = bh >> 4, h = bh & 15;
    const int q0 = blockIdx.x * 64;
    const int i0 = q0 + wave * 16;

    const _Float16* Qb = Qf + (bh * SEQ + i0) * HDIM;
    const _Float16* Kb = Kf + bh * SEQ * HDIM;
    const _Float16* Vb = Vt + bh * HDIM * SEQ;

    const int half8 = (lane >> 4) << 3;
    const int nn = lane & 15;

    // block-level staged key window [klo_blk, q0+64)
    int klo_blk = q0 - (WINDOW - 1);
    if (klo_blk < 0) klo_blk = 0;
    klo_blk &= ~31;
    const int span = q0 + 64 - klo_blk;         // multiple of 32, <= KWIN

    // ---- async stage K and V windows into LDS (ASYNCcnt path) ----
    {
        const int t = threadIdx.x;              // 0..127
        // K rows [klo_blk, klo_blk+span), contiguous: span*128 bytes in 16B chunks
        const int kChunks = span * 8;
        const uint64_t gk = (uint64_t)(uintptr_t)(Kb + klo_blk * HDIM);
        const uint32_t lk = (uint32_t)(uintptr_t)(Ks);
        for (int c = t; c < kChunks; c += 128) {
            uint32_t lds = lk + (uint32_t)c * 16u;
            uint64_t ga  = gk + (uint64_t)c * 16u;
            asm volatile("global_load_async_to_lds_b128 %0, %1, off"
                         :: "v"(lds), "v"(ga) : "memory");
        }
        // V^T rows d=0..63, cols [klo_blk, klo_blk+span): span*2 bytes per row
        const int rowChunks = span >> 3;        // 16B chunks per row
        const int vChunks = HDIM * rowChunks;
        for (int c = t; c < vChunks; c += 128) {
            const int d = c / rowChunks;
            const int w = c - d * rowChunks;
            uint32_t lds = (uint32_t)(uintptr_t)(Vs + d * KWIN + w * 8);
            uint64_t ga  = (uint64_t)(uintptr_t)(Vb + d * SEQ + klo_blk + w * 8);
            asm volatile("global_load_async_to_lds_b128 %0, %1, off"
                         :: "v"(lds), "v"(ga) : "memory");
        }
        asm volatile("s_wait_asynccnt 0" ::: "memory");
    }

    // Q A-fragments (reused for qE and QK^T) -- loaded after the async staging completes
    v16h aq0 = load_a_frag(Qb, HDIM);
    v16h aq1 = load_a_frag(Qb + 32, HDIM);

    // Phase 1: qE[i_local][r] = Q . rel_emb^T  via WMMA -> LDS
#pragma unroll
    for (int nb = 0; nb < RELPAD / 16; ++nb) {
        v16h b0 = load_b_frag(Relh + nb * 16 * HDIM, HDIM);
        v16h b1 = load_b_frag(Relh + nb * 16 * HDIM + 32, HDIM);
        v8f e = {};
        e = wmma_f16(aq0, b0, e);
        e = wmma_f16(aq1, b1, e);
#pragma unroll
        for (int v = 0; v < 8; ++v)
            qE[(wave * 16 + half8 + v) * RELPAD + nb * 16 + nn] = e[v];
    }
    __syncthreads();    // covers both qE stores and the async K/V staging

    float mstate[8], lstate[8];
#pragma unroll
    for (int v = 0; v < 8; ++v) { mstate[v] = -1e30f; lstate[v] = 0.0f; }
    v8f o[4] = {};

    _Float16* pl = pbuf + wave * (16 * 32);

    int klo = i0 - (WINDOW - 1);
    if (klo < 0) klo = 0;
    klo &= ~31;
    const int khi = i0 + 16;

    for (int jt = klo; jt < khi; jt += 32) {
        const int jl = jt - klo_blk;            // LDS-relative key offset
        // S = Q K^T for 32 keys, B-fragments from LDS
        const _Float16* Kt = Ks + jl * HDIM;
        v16h bk00 = load_b_frag(Kt, HDIM);
        v16h bk01 = load_b_frag(Kt + 32, HDIM);
        v16h bk10 = load_b_frag(Kt + 16 * HDIM, HDIM);
        v16h bk11 = load_b_frag(Kt + 16 * HDIM + 32, HDIM);
        v8f s0 = {}, s1 = {};
        s0 = wmma_f16(aq0, bk00, s0);
        s0 = wmma_f16(aq1, bk01, s0);
        s1 = wmma_f16(aq0, bk10, s1);
        s1 = wmma_f16(aq1, bk11, s1);

        float sv0[8], sv1[8], rmax[8];
#pragma unroll
        for (int v = 0; v < 8; ++v) {
            const int i = i0 + half8 + v;
            const float* qErow = &qE[(i - q0) * RELPAD];
            const int j0 = jt + nn;
            const int j1 = jt + 16 + nn;
            int r0 = j0 - i; r0 = r0 < -MAXREL ? -MAXREL : (r0 > MAXREL ? MAXREL : r0);
            int r1 = j1 - i; r1 = r1 < -MAXREL ? -MAXREL : (r1 > MAXREL ? MAXREL : r1);
            float x0 = (s0[v] + qErow[r0 + MAXREL]) * 0.125f;
            float x1 = (s1[v] + qErow[r1 + MAXREL]) * 0.125f;
            if (j0 > i || (i - j0) >= WINDOW) x0 = -1e30f;
            if (j1 > i || (i - j1) >= WINDOW) x1 = -1e30f;
            sv0[v] = x0; sv1[v] = x1;
            rmax[v] = fmaxf(x0, x1);
        }
        // row-max across the 16 lanes holding this row's columns
#pragma unroll
        for (int v = 0; v < 8; ++v)
#pragma unroll
            for (int msk = 1; msk < 16; msk <<= 1)
                rmax[v] = fmaxf(rmax[v], __shfl_xor(rmax[v], msk, 32));

#pragma unroll
        for (int v = 0; v < 8; ++v) {
            const float mnew = fmaxf(mstate[v], rmax[v]);
            const float c = __expf(mstate[v] - mnew);
            const float p0 = __expf(sv0[v] - mnew);
            const float p1 = __expf(sv1[v] - mnew);
            mstate[v] = mnew;
            float rs = p0 + p1;
#pragma unroll
            for (int msk = 1; msk < 16; msk <<= 1)
                rs += __shfl_xor(rs, msk, 32);
            lstate[v] = lstate[v] * c + rs;
#pragma unroll
            for (int db = 0; db < 4; ++db) o[db][v] *= c;
            pl[(half8 + v) * 32 + nn]      = (_Float16)p0;
            pl[(half8 + v) * 32 + 16 + nn] = (_Float16)p1;
        }
        asm volatile("s_wait_dscnt 0" ::: "memory");     // C-layout -> A-layout via LDS

        v16h ap  = load_a_frag(pl, 32);
        v16h bv0 = load_b_frag(Vs + 0  * KWIN + jl, KWIN);
        v16h bv1 = load_b_frag(Vs + 16 * KWIN + jl, KWIN);
        v16h bv2 = load_b_frag(Vs + 32 * KWIN + jl, KWIN);
        v16h bv3 = load_b_frag(Vs + 48 * KWIN + jl, KWIN);
        o[0] = wmma_f16(ap, bv0, o[0]);
        o[1] = wmma_f16(ap, bv1, o[1]);
        o[2] = wmma_f16(ap, bv2, o[2]);
        o[3] = wmma_f16(ap, bv3, o[3]);
    }

#pragma unroll
    for (int db = 0; db < 4; ++db)
#pragma unroll
        for (int v = 0; v < 8; ++v) {
            const int i = i0 + half8 + v;
            const float inv = 1.0f / lstate[v];
            Oh[(b * SEQ + i) * DMODEL + h * HDIM + db * 16 + nn] =
                (_Float16)(o[db][v] * inv);
        }
}

// ---------------- output projection GEMM ----------------
// out[m][n] = sum_k Oh[m][k] * Wout[n][k];  M=4096, N=1024, K=1024, f32 out
__global__ __launch_bounds__(256)
void k_out_gemm(const _Float16* __restrict__ Oh, const _Float16* __restrict__ Wh,
                float* __restrict__ out) {
    const int lane = threadIdx.x & 31;
    const int wave = threadIdx.x >> 5;
    const int mw = wave & 1, nw = wave >> 1;
    const int m0 = blockIdx.y * 64 + mw * 32;
    const int n0 = blockIdx.x * 128 + nw * 32;
    const _Float16* Ab = Oh + m0 * DMODEL;
    const _Float16* Bb = Wh + n0 * DMODEL;

    v8f acc[2][2] = {};
    for (int k = 0; k < DMODEL; k += 32) {
        if (k + 32 < DMODEL) {
            __builtin_prefetch(Ab + k + 32, 0, 1);
            __builtin_prefetch(Bb + k + 32, 0, 1);
        }
        v16h a0 = load_a_frag(Ab + k, DMODEL);
        v16h a1 = load_a_frag(Ab + 16 * DMODEL + k, DMODEL);
        v16h b0 = load_b_frag(Bb + k, DMODEL);
        v16h b1 = load_b_frag(Bb + 16 * DMODEL + k, DMODEL);
        acc[0][0] = wmma_f16(a0, b0, acc[0][0]);
        acc[0][1] = wmma_f16(a0, b1, acc[0][1]);
        acc[1][0] = wmma_f16(a1, b0, acc[1][0]);
        acc[1][1] = wmma_f16(a1, b1, acc[1][1]);
    }

    const int half8 = (lane >> 4) << 3;
    const int nn = lane & 15;
#pragma unroll
    for (int mi = 0; mi < 2; ++mi)
#pragma unroll
        for (int ni = 0; ni < 2; ++ni)
#pragma unroll
            for (int v = 0; v < 8; ++v) {
                const int m = m0 + mi * 16 + half8 + v;
                const int n = n0 + ni * 16 + nn;
                out[m * DMODEL + n] = acc[mi][ni][v];
            }
}

// ---------------- launcher ----------------
extern "C" void kernel_launch(void* const* d_in, const int* in_sizes, int n_in,
                              void* d_out, int out_size, void* d_ws, size_t ws_size,
                              hipStream_t stream) {
    (void)in_sizes; (void)n_in; (void)out_size; (void)ws_size;
    const float* x    = (const float*)d_in[0];
    const float* wqkv = (const float*)d_in[1];
    const float* wout = (const float*)d_in[2];
    const float* rel  = (const float*)d_in[3];
    float* out = (float*)d_out;

    char* ws = (char*)d_ws;
    const size_t MB = 1024u * 1024u;
    _Float16* Xh   = (_Float16*)(ws + 0 * MB);   // 4096x1024 -> 8 MB
    _Float16* Wqh  = (_Float16*)(ws + 8 * MB);   // 3072x1024 -> 6 MB
    _Float16* Woh  = (_Float16*)(ws + 14 * MB);  // 1024x1024 -> 2 MB
    _Float16* Relh = (_Float16*)(ws + 16 * MB);  // 144x64
    _Float16* Qf   = (_Float16*)(ws + 17 * MB);  // [64][1024][64] -> 8 MB
    _Float16* Kf   = (_Float16*)(ws + 25 * MB);  // 8 MB
    _Float16* Vt   = (_Float16*)(ws + 33 * MB);  // [64][64][1024] -> 8 MB
    _Float16* Oh   = (_Float16*)(ws + 41 * MB);  // 4096x1024 -> 8 MB

    const int nx = NB * SEQ * DMODEL;            // 4194304
    const int nwq = 3 * DMODEL * DMODEL;         // 3145728
    const int nwo = DMODEL * DMODEL;             // 1048576
    k_cvt_f16<<<(nx + 255) / 256, 256, 0, stream>>>(x, Xh, nx);
    k_cvt_f16<<<(nwq + 255) / 256, 256, 0, stream>>>(wqkv, Wqh, nwq);
    k_cvt_f16<<<(nwo + 255) / 256, 256, 0, stream>>>(wout, Woh, nwo);
    k_rel_pad<<<(RELPAD * HDIM + 255) / 256, 256, 0, stream>>>(rel, Relh);

    k_qkv_gemm<<<dim3(3 * DMODEL / 128, NB * SEQ / 64), 256, 0, stream>>>(Xh, Wqh, Qf, Kf, Vt);
    k_attn<<<dim3(SEQ / 64, NB * NHEADS), 128, 0, stream>>>(Qf, Kf, Vt, Relh, Oh);
    k_out_gemm<<<dim3(DMODEL / 128, NB * SEQ / 64), 256, 0, stream>>>(Oh, Woh, out);
}
